// MultiheadAttention_72842645340368
// MI455X (gfx1250) — matrix operand order repaired and checked by reading, not compiled, not run on hardware
//
#include <hip/hip_runtime.h>

// ---------------- problem constants ----------------
#define Bn 4
#define Sn 2048
#define Dn 1024
#define Hn 16
#define HDn 64
#define Mn (Bn * Sn)           // 8192 rows

typedef __attribute__((ext_vector_type(16))) __bf16 bf16x16;
typedef __attribute__((ext_vector_type(4)))  __bf16 bf16x4;
typedef __attribute__((ext_vector_type(8)))  float  f32x8;
typedef __attribute__((ext_vector_type(4)))  unsigned int u32x4;
typedef __attribute__((ext_vector_type(4)))  int i32x4;
typedef __attribute__((ext_vector_type(8)))  int i32x8;

// Load a 16-element bf16 fragment as two 16B chunks (A/B fragment halves are
// contiguous runs of 8 K-values per the CDNA5 16-bit WMMA VGPR layouts).
static __device__ __forceinline__ bf16x16 ld_pair(const __bf16* p0, const __bf16* p1) {
  union { u32x4 q[2]; bf16x16 v; } u;
  u.q[0] = *(const u32x4*)p0;
  u.q[1] = *(const u32x4*)p1;
  return u.v;
}

static __device__ __forceinline__ f32x8 wmma_bf16(bf16x16 a, bf16x16 b, f32x8 c) {
  // (neg_a, A, neg_b, B, c_mod, C, reuse_a, reuse_b)
  return __builtin_amdgcn_wmma_f32_16x16x32_bf16(false, a, false, b, (short)0, c,
                                                 false, false);
}

// ---- Tensor Data Mover: DMA one 64-row x 32-col bf16 tile (row stride =
// row_stride_elems) from global into LDS, writing rows padded 64B+16B so the
// LDS image has a 40-bf16 row stride (bank-conflict padding done by the TDM).
// Descriptor per CDNA5 ISA ch.8: D# group0/group1; groups 2/3 unused (2-D).
static __device__ __forceinline__ void tdm_load_tile_64x32(const __bf16* gptr,
                                                           void* lds,
                                                           int row_stride_elems) {
  const unsigned lds_addr = (unsigned)(size_t)lds;               // low 32 bits = LDS offset
  const unsigned long long ga = (unsigned long long)(size_t)gptr;
  u32x4 g0;
  g0[0] = 1u;                                                    // count=1, no gather
  g0[1] = lds_addr;                                              // lds_addr [63:32]
  g0[2] = (unsigned)ga;                                          // global_addr lo
  g0[3] = (unsigned)((ga >> 32) & 0x01FFFFFFu) | 0x80000000u;    // ga hi | type=2
  i32x8 g1;
  g1[0] = (1 << 16)      // data_size = 1 -> 2 bytes
        | (1 << 20)      // pad_enable
        | (3 << 22)      // pad_interval: 16 DWORDs = 64B between pads
        | (3 << 25);     // pad_amount: 4 DWORDs = 16B of padding
  g1[1] = (row_stride_elems & 0xFFFF) << 16;                     // abar=0 | tensor_dim0 lo
  g1[2] = ((row_stride_elems >> 16) & 0xFFFF) | (64 << 16);      // dim0 hi | tensor_dim1 lo (64)
  g1[3] = (32 << 16);                                            // dim1 hi=0 | tile_dim0=32
  g1[4] = 64;                                                    // tile_dim1=64, tile_dim2=0
  g1[5] = row_stride_elems;                                      // tensor_dim0_stride lo32
  g1[6] = 0;                                                     // stride hi | dim1_stride lo
  g1[7] = 0;
  const i32x4 z4 = {0, 0, 0, 0};
#if defined(__clang_major__) && __clang_major__ >= 23
  const i32x8 z8 = {0, 0, 0, 0, 0, 0, 0, 0};
  __builtin_amdgcn_tensor_load_to_lds(g0, g1, z4, z4, z8, 0);
#else
  __builtin_amdgcn_tensor_load_to_lds(g0, g1, z4, z4, 0);
#endif
}

// ---------------- fp32 -> bf16 cast ----------------
__global__ void cast_f32_bf16(const float* __restrict__ src, __bf16* __restrict__ dst, int n4) {
  int i = blockIdx.x * blockDim.x + threadIdx.x;
  if (i < n4) {
    const float4 f = ((const float4*)src)[i];
    bf16x4 o;
    o.x = (__bf16)f.x; o.y = (__bf16)f.y; o.z = (__bf16)f.z; o.w = (__bf16)f.w;
    ((bf16x4*)dst)[i] = o;
  }
}

// ---------------- tiled bf16 GEMM: C = A(MxK) * W(NxK)^T ----------------
// 128 threads = 4 waves; workgroup tile 64x64; each wave owns a 32x32 subtile
// (2x2 WMMA frags). K stepped by 32; tiles DMA'd into double-buffered LDS by
// the Tensor Data Mover (wave 0 issues, TENSORcnt + barrier publishes).
// EPI: 0 = RoPE -> bf16 [b,h,s,d]   1 = bf16 [b,h,d,s] (V transposed)
//      2 = fp32 row-major (final projection)
template <int EPI>
__global__ __launch_bounds__(128)
void gemm_bf16(const __bf16* __restrict__ A, const __bf16* __restrict__ W,
               void* __restrict__ dst, int M, int N, int K) {
  __shared__ __bf16 sA[2][64 * 40];   // 40-bf16 row stride (TDM pads 64B+16B)
  __shared__ __bf16 sW[2][64 * 40];

  const int tid  = threadIdx.x;
  const int lane = tid & 31;
  const int wave = tid >> 5;
  const int wy = wave >> 1, wx = wave & 1;
  const int lm = lane & 15, kh = lane >> 4;
  const int m0 = blockIdx.x * 64;
  const int n0 = blockIdx.y * 64;

  f32x8 acc[2][2] = {};
  const int nk = K / 32;

  // prologue: DMA tile 0 into buffer 0
  if (wave == 0) {
    tdm_load_tile_64x32(A + (size_t)m0 * K, sA[0], K);
    tdm_load_tile_64x32(W + (size_t)n0 * K, sW[0], K);
    __builtin_amdgcn_s_wait_tensorcnt(0);
  }
  __syncthreads();

  for (int kt = 0; kt < nk; ++kt) {
    const int cur = kt & 1, nxt = cur ^ 1;
    if (kt + 1 < nk && wave == 0) {   // overlap next tile's DMA with compute
      tdm_load_tile_64x32(A + (size_t)m0 * K + (kt + 1) * 32, sA[nxt], K);
      tdm_load_tile_64x32(W + (size_t)n0 * K + (kt + 1) * 32, sW[nxt], K);
    }

    bf16x16 af[2], bf[2];
#pragma unroll
    for (int fa = 0; fa < 2; ++fa) {
      const __bf16* r = sA[cur] + (wy * 32 + fa * 16 + lm) * 40;
      af[fa] = ld_pair(r + 8 * kh, r + 16 + 8 * kh);     // A 16x32 layout
    }
#pragma unroll
    for (int fb = 0; fb < 2; ++fb) {
      const __bf16* r = sW[cur] + (wx * 32 + fb * 16 + lm) * 40;
      bf[fb] = ld_pair(r + 8 * kh, r + 16 + 8 * kh);     // B 32x16 layout
    }
#pragma unroll
    for (int fa = 0; fa < 2; ++fa)
#pragma unroll
      for (int fb = 0; fb < 2; ++fb)
        acc[fa][fb] = wmma_bf16(af[fa], bf[fb], acc[fa][fb]);

    if (wave == 0) __builtin_amdgcn_s_wait_tensorcnt(0);
    __syncthreads();   // publish next buffer + close read window on current
  }

  // -------- epilogue: D layout = lane holds col n=lm, row m = r + 8*kh --------
#pragma unroll
  for (int fa = 0; fa < 2; ++fa) {
#pragma unroll
    for (int fb = 0; fb < 2; ++fb) {
#pragma unroll
      for (int r = 0; r < 8; ++r) {
        float v = acc[fa][fb][r];
        const int m = m0 + wy * 32 + fa * 16 + r + 8 * kh;  // b*S + s
        const int n = n0 + wx * 32 + fb * 16 + lm;          // h*64 + d
        if (EPI == 2) {
          ((float*)dst)[(size_t)m * N + n] = v;
        } else {
          const int b = m >> 11, s = m & (Sn - 1);
          const int h = n >> 6,  d = n & 63;
          if (EPI == 0) {  // fused RoPE; pair column sits in lane^1, same r
            const float partner = __shfl_xor(v, 1);
            const float freq = __powf(10000.0f, -(float)(d & ~1) * (1.0f / 64.0f));
            float sn, cs;
            __sincosf((float)s * freq, &sn, &cs);
            v = (d & 1) ? fmaf(partner, sn, v * cs) : (v * cs - partner * sn);
            ((__bf16*)dst)[(((size_t)(b * Hn + h)) * Sn + s) * 64 + d] = (__bf16)v;
          } else {         // EPI == 1: V stored transposed [b,h,d,s]
            ((__bf16*)dst)[((size_t)(b * Hn + h) * 64 + d) * Sn + s] = (__bf16)v;
          }
        }
      }
    }
  }
}

// ---------------- flash attention: 1 wave per (b,h, 16-query tile) ----------------
// Q,K: bf16 [b,h,s,64]; Vt: bf16 [b,h,d,s]; out: bf16 [b,s,h*64+d]
//
// Softmax runs in the A-fragment layout: a P-tile row lives in exactly two lanes
// (lm, lm+16), so row max / row sum need one __shfl_xor(.,16) each, and the exp'd
// probabilities are produced directly in A-fragment element order (the PV A-frag
// is assembled in registers -- no second LDS pass).
__global__ __launch_bounds__(32)
void flash_attn(const __bf16* __restrict__ Q, const __bf16* __restrict__ Kp,
                const __bf16* __restrict__ Vt, __bf16* __restrict__ Aout) {
  __shared__ float lds_s[16 * 40];  // raw score tile, D-layout -> A-layout bounce

  const int lane = threadIdx.x;
  const int lm = lane & 15, kh = lane >> 4;
  const int bh = blockIdx.x;            // b*H + h
  const int q0 = blockIdx.y * 16;
  const int b = bh >> 4, h = bh & 15;

  const __bf16* Qp = Q + (size_t)bh * Sn * 64;
  const __bf16* Kb = Kp + (size_t)bh * Sn * 64;
  const __bf16* Vb = Vt + (size_t)bh * 64 * Sn;

  // Q fragments (rows q0..q0+15, d split 0-31 / 32-63), loaded once
  const __bf16* qrow = Qp + (size_t)(q0 + lm) * 64;
  const bf16x16 qa0 = ld_pair(qrow + 8 * kh, qrow + 16 + 8 * kh);
  const bf16x16 qa1 = ld_pair(qrow + 32 + 8 * kh, qrow + 48 + 8 * kh);

  float mrow = -1e30f;   // running row max (row = q0+lm, replicated in lane pair)
  float lrow = 0.0f;     // running row sum
  f32x8 o[4] = {};

  const float scale = 0.125f;  // 1/sqrt(64)
  const int nkt = (q0 + 15) / 32 + 1;

  for (int kt = 0; kt < nkt; ++kt) {
    const int kb = kt * 32;

    // ---- QK^T: batch all 4 K B-frags, then 4 WMMAs ----
    bf16x16 kf[4];
#pragma unroll
    for (int nt = 0; nt < 2; ++nt) {
      const __bf16* krow = Kb + (size_t)(kb + nt * 16 + lm) * 64;
      kf[nt * 2]     = ld_pair(krow + 8 * kh,      krow + 16 + 8 * kh);       // d 0..31
      kf[nt * 2 + 1] = ld_pair(krow + 32 + 8 * kh, krow + 48 + 8 * kh);       // d 32..63
    }
    if (kt + 1 < nkt) {
      __builtin_prefetch(Kb + (size_t)(kb + 32 + lane) * 64, 0, 0);
      __builtin_prefetch(Vb + (size_t)(lane * 2) * Sn + kb + 32, 0, 0);
    }
    f32x8 sc[2] = {};
    sc[0] = wmma_bf16(qa0, kf[0], sc[0]);
    sc[0] = wmma_bf16(qa1, kf[1], sc[0]);
    sc[1] = wmma_bf16(qa0, kf[2], sc[1]);
    sc[1] = wmma_bf16(qa1, kf[3], sc[1]);

    // ---- dump raw scores (D layout) to LDS ----
#pragma unroll
    for (int nt = 0; nt < 2; ++nt)
#pragma unroll
      for (int r = 0; r < 8; ++r)
        lds_s[(r + 8 * kh) * 40 + nt * 16 + lm] = sc[nt][r];
    __syncthreads();

    // ---- read back in A-fragment element order: row = lm ----
    union { u32x4 q[4]; float f[16]; } uf;
    {
      const float* srow = lds_s + lm * 40;
      uf.q[0] = *(const u32x4*)(srow + 8 * kh);
      uf.q[1] = *(const u32x4*)(srow + 8 * kh + 4);
      uf.q[2] = *(const u32x4*)(srow + 16 + 8 * kh);
      uf.q[3] = *(const u32x4*)(srow + 16 + 8 * kh + 4);
    }
    __syncthreads();  // reads done; safe for next iteration's stores

    // ---- lane-local scale + causal mask + online softmax ----
    const int q = q0 + lm;
    const bool needmask = (kb + 31 > q0);
    float pv[16];
    float tmax = -1e30f;
#pragma unroll
    for (int i = 0; i < 16; ++i) {
      // A-frag element i -> local key index (two contiguous runs of 8)
      const int klocal = (i < 8) ? (8 * kh + i) : (8 + 8 * kh + i);
      float v = uf.f[i] * scale;
      if (needmask && (kb + klocal > q)) v = -1e30f;
      pv[i] = v;
      tmax = fmaxf(tmax, v);
    }
    tmax = fmaxf(tmax, __shfl_xor(tmax, 16));  // combine the lane pair
    const float mnew = fmaxf(mrow, tmax);
    const float cf = __expf(mrow - mnew);
    mrow = mnew;

    float psum = 0.0f;
    bf16x16 pa;
#pragma unroll
    for (int i = 0; i < 16; ++i) {
      const float e = __expf(pv[i] - mnew);
      psum += e;
      pa[i] = (__bf16)e;   // PV A-fragment assembled in registers
    }
    psum += __shfl_xor(psum, 16);
    lrow = lrow * cf + psum;

    // ---- rescale O accumulators (D layout: row r+8*kh needs cf from lane r+8*kh)
#pragma unroll
    for (int r = 0; r < 8; ++r) {
      const float cfm = __shfl(cf, r + 8 * kh);
#pragma unroll
      for (int j = 0; j < 4; ++j) o[j][r] = o[j][r] * cfm;
    }

    // ---- PV: batch all 4 V B-frags, then 4 WMMAs ----
    bf16x16 vf[4];
#pragma unroll
    for (int j = 0; j < 4; ++j) {
      const __bf16* vrow = Vb + (size_t)(j * 16 + lm) * Sn + kb;
      vf[j] = ld_pair(vrow + 8 * kh, vrow + 16 + 8 * kh);
    }
#pragma unroll
    for (int j = 0; j < 4; ++j) o[j] = wmma_bf16(pa, vf[j], o[j]);
  }

  // ---- epilogue: divide by row sum, store bf16 [b,s,h*64+d] ----
#pragma unroll
  for (int r = 0; r < 8; ++r) {
    const float lsum = __shfl(lrow, r + 8 * kh);
    const float inv = 1.0f / lsum;
    const int qq = q0 + r + 8 * kh;
#pragma unroll
    for (int j = 0; j < 4; ++j) {
      const int d = j * 16 + lm;
      Aout[((size_t)b * Sn + qq) * Dn + h * 64 + d] = (__bf16)(o[j][r] * inv);
    }
  }
}

// ---------------- host-side pipeline ----------------
extern "C" void kernel_launch(void* const* d_in, const int* in_sizes, int n_in,
                              void* d_out, int out_size, void* d_ws, size_t ws_size,
                              hipStream_t stream) {
  const float* x  = (const float*)d_in[0];
  const float* Wq = (const float*)d_in[1];
  const float* Wk = (const float*)d_in[2];
  const float* Wv = (const float*)d_in[3];
  const float* Wo = (const float*)d_in[4];
  float* out = (float*)d_out;

  char* ws = (char*)d_ws;
  const size_t EX = (size_t)Mn * Dn;   // 8192*1024 elements
  const size_t EW = (size_t)Dn * Dn;   // 1024*1024 elements
  __bf16* xb   = (__bf16*)(ws);
  __bf16* Wqb  = (__bf16*)(ws + 2 * EX);
  __bf16* Wkb  = (__bf16*)(ws + 2 * (EX + EW));
  __bf16* Wvb  = (__bf16*)(ws + 2 * (EX + 2 * EW));
  __bf16* Wob  = (__bf16*)(ws + 2 * (EX + 3 * EW));
  __bf16* Qb   = (__bf16*)(ws + 2 * (EX + 4 * EW));
  __bf16* Kbuf = (__bf16*)(ws + 2 * (2 * EX + 4 * EW));
  __bf16* Vtb  = (__bf16*)(ws + 2 * (3 * EX + 4 * EW));
  __bf16* Ab   = (__bf16*)(ws + 2 * (4 * EX + 4 * EW));

  // 1) bf16 casts
  cast_f32_bf16<<<(int)((EX / 4 + 255) / 256), 256, 0, stream>>>(x, xb, (int)(EX / 4));
  cast_f32_bf16<<<(int)((EW / 4 + 255) / 256), 256, 0, stream>>>(Wq, Wqb, (int)(EW / 4));
  cast_f32_bf16<<<(int)((EW / 4 + 255) / 256), 256, 0, stream>>>(Wk, Wkb, (int)(EW / 4));
  cast_f32_bf16<<<(int)((EW / 4 + 255) / 256), 256, 0, stream>>>(Wv, Wvb, (int)(EW / 4));
  cast_f32_bf16<<<(int)((EW / 4 + 255) / 256), 256, 0, stream>>>(Wo, Wob, (int)(EW / 4));

  // 2) projections (RoPE fused for Q/K, V stored transposed)
  dim3 gg(Mn / 64, Dn / 64);
  gemm_bf16<0><<<gg, 128, 0, stream>>>(xb, Wqb, Qb,   Mn, Dn, Dn);
  gemm_bf16<0><<<gg, 128, 0, stream>>>(xb, Wkb, Kbuf, Mn, Dn, Dn);
  gemm_bf16<1><<<gg, 128, 0, stream>>>(xb, Wvb, Vtb,  Mn, Dn, Dn);

  // 3) causal flash attention
  flash_attn<<<dim3(Bn * Hn, Sn / 16), 32, 0, stream>>>(Qb, Kbuf, Vtb, Ab);

  // 4) output projection -> fp32
  gemm_bf16<2><<<gg, 128, 0, stream>>>(Ab, Wob, out, Mn, Dn, Dn);
}
